// PGTSTGraphNeuralCDE_12206297055717
// MI455X (gfx1250) — compile-verified, matrix operand order checked by reading, not run on hardware
//
#include <hip/hip_runtime.h>

// ---------------------------------------------------------------------------
// Neural CDE (z-channel only; h channel is dead w.r.t. the output).
// One wave = 16 nodes, full 255-step RK4 loop in-kernel.
// Weights as WMMA A-operand, activations as B-operand: layer1 C-tile IS the
// layer2 B-fragment (register-only handoff); einsum is an in-lane dot product;
// RK4 state (y, k-accumulator) lives in per-lane registers (stage-invariant
// lane ownership). dX via global_load_async_to_lds_b128 (ASYNCcnt).
// dt cancels algebraically out of the RK4 recurrence (no divides).
// ---------------------------------------------------------------------------

#define N_NODES 10000
#define T_STEPS 256
#define H_DIM   32

typedef __attribute__((ext_vector_type(16))) __bf16 v16bf;
typedef __attribute__((ext_vector_type(8)))  __bf16 v8bf;
typedef __attribute__((ext_vector_type(8)))  float  v8f;
typedef __attribute__((ext_vector_type(4)))  int    v4i;

#define WAVES_PER_WG 4
#define THREADS (WAVES_PER_WG * 32)
#define TILES   ((N_NODES + 15) / 16)   // 625

// ---- dynamic-LDS layout (bytes) ----
#define OFF_W2   0
#define SZ_W2    (2*16*32*16*2)          // [kt][ct][lane][16] bf16 = 32768
#define OFF_W1   (OFF_W2 + SZ_W2)
#define SZ_W1    (4*32*16*2)             // [nt][lane][16] bf16 = 4096
#define OFF_B1   (OFF_W1 + SZ_W1)
#define SZ_B1    (64*4)
#define OFF_B2   (OFF_B1 + SZ_B1)
#define SZ_B2    (256*4)
#define OFF_WAVE (OFF_B2 + SZ_B2)        // 38144; per-wave blocks follow
#define PW_YZ    0                        // float [16][32] (decoder handoff)
#define PW_ZA    2048                     // bf16  [16][32]
#define PW_XRA   3072                     // float [16][8]  x_coeffs row t
#define PW_XRB   3584                     // float [16][8]  x_coeffs row t+1
#define PW_SIZE  4096
#define SMEM_TOTAL (OFF_WAVE + WAVES_PER_WG * PW_SIZE)  // 54528 bytes

#if defined(__gfx1250__) && __has_builtin(__builtin_amdgcn_global_load_async_to_lds_b128)
#define HAS_ASYNC_LDS 1
#else
#define HAS_ASYNC_LDS 0
#endif

#if HAS_ASYNC_LDS
typedef v4i __attribute__((address_space(1))) g_v4i;   // global (AS1)
typedef v4i __attribute__((address_space(3))) l_v4i;   // LDS (AS3)
#endif

__device__ __forceinline__ void wave_lds_sync() {
  __builtin_amdgcn_fence(__ATOMIC_RELEASE, "wavefront");
  __builtin_amdgcn_wave_barrier();
  __builtin_amdgcn_fence(__ATOMIC_ACQUIRE, "wavefront");
}

// Single-instruction relu: v_med3(x, 0, +inf).
__device__ __forceinline__ float relu1(float x) {
  return __builtin_amdgcn_fmed3f(x, 0.0f, __builtin_inff());
}

// Issue one 16B-per-lane row copy (128 floats) global -> per-wave LDS buffer.
__device__ __forceinline__ void row_load(const float* gsrc, float* ldst, int lane) {
#if HAS_ASYNC_LDS
  const float* g = gsrc + lane * 4;
  float*       l = ldst + lane * 4;
  // AS1 addr == flat addr numerically; AS3 offset == flat addr low 32 bits.
  __builtin_amdgcn_global_load_async_to_lds_b128(
      (g_v4i*)(uintptr_t)g,
      (l_v4i*)(unsigned)(uintptr_t)l,
      0, 0);
#else
  const float4 v = *(const float4*)(gsrc + lane * 4);
  *(float4*)(ldst + lane * 4) = v;
#endif
}

__device__ __forceinline__ void wait_rows() {
#if HAS_ASYNC_LDS
#if __has_builtin(__builtin_amdgcn_s_wait_asynccnt)
  __builtin_amdgcn_s_wait_asynccnt(0);
#else
  asm volatile("s_wait_asynccnt 0x0" ::: "memory");
#endif
#endif
  wave_lds_sync();
}

// 16x32 bf16 fragment from a row-major LDS matrix (A- and B-fragments share
// the same per-lane mapping in this scheme):
// lane l (r=l&15, s=l>>4): elems 0..7 -> K=s*8+e; elems 8..15 -> K=16+s*8+(e-8).
__device__ __forceinline__ v16bf load_frag(const __bf16* rm, int stride, int lane) {
  const int r = lane & 15, s = lane >> 4;
  const __bf16* base = rm + r * stride + s * 8;
  v8bf lo = *(const v8bf*)(base);
  v8bf hi = *(const v8bf*)(base + 16);
  return __builtin_shufflevector(lo, hi, 0,1,2,3,4,5,6,7,
                                         8,9,10,11,12,13,14,15);
}

__global__ void __launch_bounds__(THREADS, 1)
ncde_rk4_wmma_kernel(const float* __restrict__ ts,   // unused: dt cancels in RK4
                     const float* __restrict__ xc,   // [T][N][8]
                     const float* __restrict__ x0,   // [N][8]
                     const float* __restrict__ ez_w0, const float* __restrict__ ez_b0,
                     const float* __restrict__ ez_w1, const float* __restrict__ ez_b1,
                     const float* __restrict__ ez_w2, const float* __restrict__ ez_b2,
                     const float* __restrict__ dc_w0, const float* __restrict__ dc_b0,
                     const float* __restrict__ dc_w1, const float* __restrict__ dc_b1,
                     const float* __restrict__ dc_w2, const float* __restrict__ dc_b2,
                     const float* __restrict__ g_w0,  const float* __restrict__ g_b0, // [64][32],[64]
                     const float* __restrict__ g_w1,  const float* __restrict__ g_b1, // [256][64],[256]
                     float* __restrict__ out)         // [N][8]
{
  (void)ts;
  extern __shared__ char dsm[];
  __bf16* W2s = (__bf16*)(dsm + OFF_W2);
  __bf16* W1s = (__bf16*)(dsm + OFF_W1);
  float*  b1s = (float*)(dsm + OFF_B1);
  float*  b2s = (float*)(dsm + OFF_B2);

  const int tid  = threadIdx.x;
  const int lane = tid & 31;
  const int wave = tid >> 5;

  // ---- swizzle loop-invariant weights into per-lane A-fragment layout ----
  // A (MxK) frag: lane l holds row n=(l&15); elems 0..7 -> k=(l>>4)*8+e,
  // elems 8..15 -> k=16+(l>>4)*8+(e-8). W stored [out][in] = [n][k].
  for (int idx = tid; idx < 4*32*16; idx += THREADS) {
    const int e = idx & 15, ln = (idx >> 4) & 31, nt = idx >> 9;
    const int n = nt*16 + (ln & 15);
    const int k = (e < 8) ? ((ln >> 4)*8 + e) : (16 + (ln >> 4)*8 + (e - 8));
    W1s[idx] = (__bf16)g_w0[n*32 + k];
  }
  for (int idx = tid; idx < 2*16*32*16; idx += THREADS) {
    const int e = idx & 15, ln = (idx >> 4) & 31;
    const int ct = (idx >> 9) & 15, kt = idx >> 13;
    const int n = ct*16 + (ln & 15);
    const int k = kt*32 + ((e < 8) ? ((ln >> 4)*8 + e)
                                   : (16 + (ln >> 4)*8 + (e - 8)));
    W2s[idx] = (__bf16)g_w1[n*64 + k];
  }
  for (int i = tid; i < 64;  i += THREADS) b1s[i] = g_b0[i];
  for (int i = tid; i < 256; i += THREADS) b2s[i] = g_b1[i];
  __syncthreads();

  const int tile = blockIdx.x * WAVES_PER_WG + wave;
  if (tile >= TILES) return;
  const int base_n = tile * 16;

  char*   pw   = dsm + OFF_WAVE + wave * PW_SIZE;
  float*  yz   = (float*)(pw + PW_YZ);
  __bf16* zA   = (__bf16*)(pw + PW_ZA);
  float*  xrA  = (float*)(pw + PW_XRA);
  float*  xrB  = (float*)(pw + PW_XRB);

  // kick off DMA of x_coeffs rows 0 and 1 (overlaps with encoder)
  row_load(xc + ((size_t)0 * N_NODES + base_n) * 8, xrA, lane);
  row_load(xc + ((size_t)1 * N_NODES + base_n) * 8, xrB, lane);

  // ---- encoder (enc_z only): lanes 0..15, one node each ----
  if (lane < 16) {
    const int n = base_n + lane;
    float x[8];
    #pragma unroll
    for (int d = 0; d < 8; ++d) x[d] = x0[n*8 + d];
    float t1[16], t2[16];
    for (int i = 0; i < 16; ++i) {
      float a = ez_b0[i];
      for (int d = 0; d < 8; ++d) a += ez_w0[i*8 + d] * x[d];
      t1[i] = relu1(a);
    }
    for (int i = 0; i < 16; ++i) {
      float a = ez_b1[i];
      for (int d = 0; d < 16; ++d) a += ez_w1[i*16 + d] * t1[d];
      t2[i] = relu1(a);
    }
    for (int j = 0; j < H_DIM; ++j) {
      float a = ez_b2[j];
      for (int d = 0; d < 16; ++d) a += ez_w2[j*16 + d] * t2[d];
      yz[lane*32 + j] = a;
      zA[lane*32 + j] = (__bf16)a;
    }
  }
  wave_lds_sync();

  const int nn    = lane & 15;   // this lane's node column
  const int shalf = lane >> 4;   // row-half of C tiles
  const int r_lo  = shalf * 8;

  // ---- hoist loop-invariant fragments / biases into registers ----
  v16bf w1f[4];
  v8f   b1c[4];
  #pragma unroll
  for (int nt = 0; nt < 4; ++nt) {
    w1f[nt] = *(const v16bf*)(W1s + (nt*32 + lane) * 16);
    b1c[nt] = *(const v8f*)(b1s + nt*16 + r_lo);
  }

  const float cnext[4] = {0.5f, 0.5f, 1.0f, 0.0f};  // coeff for next stage input
  const float wacc[4]  = {1.0f, 2.0f, 2.0f, 1.0f};  // RK4 accumulation weights

  // RK4 state in registers: lane owns (node nn, h = 2*ct + shalf), ct = 0..15.
  float yreg[16], kreg[16];
  #pragma unroll
  for (int ct = 0; ct < 16; ++ct) {
    yreg[ct] = yz[nn*32 + 2*ct + shalf];
    kreg[ct] = 0.f;
  }

  for (int t = 0; t < T_STEPS - 1; ++t) {
    wait_rows();   // rows t (xrA) and t+1 (xrB) now resident in LDS

    // dX for this lane's node (dt cancels out of RK4; no divide needed)
    float dxv[8];
    {
      const float4 a0 = *(const float4*)(xrA + nn*8);
      const float4 a1 = *(const float4*)(xrA + nn*8 + 4);
      const float4 b0 = *(const float4*)(xrB + nn*8);
      const float4 b1 = *(const float4*)(xrB + nn*8 + 4);
      dxv[0] = b0.x - a0.x; dxv[1] = b0.y - a0.y;
      dxv[2] = b0.z - a0.z; dxv[3] = b0.w - a0.w;
      dxv[4] = b1.x - a1.x; dxv[5] = b1.y - a1.y;
      dxv[6] = b1.z - a1.z; dxv[7] = b1.w - a1.w;
    }
    // all dxv in registers; safe to overwrite buffers with next rows
    asm volatile("s_wait_dscnt 0x0" ::: "memory");
    if (t < T_STEPS - 2) {
      row_load(xc + ((size_t)(t+1) * N_NODES + base_n) * 8, xrA, lane);
      row_load(xc + ((size_t)(t+2) * N_NODES + base_n) * 8, xrB, lane);
      if (t + 3 < T_STEPS)
        __builtin_prefetch(xc + ((size_t)(t+3) * N_NODES + base_n) * 8 + lane * 4, 0, 1);
    }

    #pragma unroll
    for (int st = 0; st < 4; ++st) {
      // ---- layer 1: C'[unit][node] = W1 x z^T + b1, relu ----
      // C' VGPR r of lane l = unit nt*16 + r_lo + r for node nn: exactly the
      // layer2 B-fragment element layout -> register-only handoff.
      v16bf zb = load_frag(zA, 32, lane);
      float rl[4][8];
      #pragma unroll
      for (int nt = 0; nt < 4; ++nt) {
        v8f c = __builtin_amdgcn_wmma_f32_16x16x32_bf16(false, w1f[nt], false, zb,
                                                        (short)0, b1c[nt],
                                                        false, false);
        #pragma unroll
        for (int r = 0; r < 8; ++r) rl[nt][r] = relu1(c[r]);
      }
      v16bf B20, B21;
      #pragma unroll
      for (int e = 0; e < 8; ++e) {
        B20[e]     = (__bf16)rl[0][e];
        B20[e + 8] = (__bf16)rl[1][e];
        B21[e]     = (__bf16)rl[2][e];
        B21[e + 8] = (__bf16)rl[3][e];
      }

      // ---- layer 2 + in-lane einsum + RK4 update (all in registers) ----
      const float cs = cnext[st];
      const float ws = wacc[st];
      for (int ct = 0; ct < 16; ++ct) {
        v16bf A0 = *(const v16bf*)(W2s + ((0*16 + ct)*32 + lane) * 16);
        v16bf A1 = *(const v16bf*)(W2s + ((16   + ct)*32 + lane) * 16);
        v8f acc = *(const v8f*)(b2s + ct*16 + r_lo);  // bias via C operand
        acc = __builtin_amdgcn_wmma_f32_16x16x32_bf16(false, A0, false, B20,
                                                      (short)0, acc, false, false);
        acc = __builtin_amdgcn_wmma_f32_16x16x32_bf16(false, A1, false, B21,
                                                      (short)0, acc, false, false);
        // acc[r] = F[node nn][(2ct+shalf)*8 + r]; dz = dot(acc, dx[nn])
        float dz = acc[0] * dxv[0];
        #pragma unroll
        for (int r = 1; r < 8; ++r) dz = fmaf(acc[r], dxv[r], dz);
        kreg[ct] = fmaf(ws, dz, kreg[ct]);
        if (st < 3) zA[nn*32 + 2*ct + shalf] = (__bf16)fmaf(cs, dz, yreg[ct]);
      }
      wave_lds_sync();
    }

    // ---- y += 1/6 * (k1 + 2k2 + 2k3 + k4); refresh zA; clear k ----
    #pragma unroll
    for (int ct = 0; ct < 16; ++ct) {
      yreg[ct] = fmaf(1.0f/6.0f, kreg[ct], yreg[ct]);
      kreg[ct] = 0.f;
      zA[nn*32 + 2*ct + shalf] = (__bf16)yreg[ct];
    }
    wave_lds_sync();
  }

  // ---- hand final z back to LDS for the decoder ----
  #pragma unroll
  for (int ct = 0; ct < 16; ++ct)
    yz[nn*32 + 2*ct + shalf] = yreg[ct];
  wave_lds_sync();

  // ---- decoder on final z: lanes 0..15, one node each ----
  if (lane < 16) {
    const int n = base_n + lane;
    float z[32], t1[16], t2[16];
    for (int j = 0; j < 32; ++j) z[j] = yz[lane*32 + j];
    for (int i = 0; i < 16; ++i) {
      float a = dc_b0[i];
      for (int d = 0; d < 32; ++d) a += dc_w0[i*32 + d] * z[d];
      t1[i] = relu1(a);
    }
    for (int i = 0; i < 16; ++i) {
      float a = dc_b1[i];
      for (int d = 0; d < 16; ++d) a += dc_w1[i*16 + d] * t1[d];
      t2[i] = relu1(a);
    }
    for (int f = 0; f < 8; ++f) {
      float a = dc_b2[f];
      for (int d = 0; d < 16; ++d) a += dc_w2[f*16 + d] * t2[d];
      out[n*8 + f] = a;
    }
  }
}

extern "C" void kernel_launch(void* const* d_in, const int* in_sizes, int n_in,
                              void* d_out, int out_size, void* d_ws, size_t ws_size,
                              hipStream_t stream) {
  (void)in_sizes; (void)n_in; (void)d_ws; (void)ws_size; (void)out_size;
  // setup_inputs() dict order, pytree leaves flattened (W,b per layer):
  // 0 ts | 1 coeffs_adj | 2 x_coeffs | 3 x0 | 4..9 enc_h | 10..15 enc_z |
  // 16..21 dec | 22..25 f_params | 26..29 g_params
  const float* ts    = (const float*)d_in[0];
  const float* xc    = (const float*)d_in[2];
  const float* x0    = (const float*)d_in[3];
  const float* ez_w0 = (const float*)d_in[10];
  const float* ez_b0 = (const float*)d_in[11];
  const float* ez_w1 = (const float*)d_in[12];
  const float* ez_b1 = (const float*)d_in[13];
  const float* ez_w2 = (const float*)d_in[14];
  const float* ez_b2 = (const float*)d_in[15];
  const float* dc_w0 = (const float*)d_in[16];
  const float* dc_b0 = (const float*)d_in[17];
  const float* dc_w1 = (const float*)d_in[18];
  const float* dc_b1 = (const float*)d_in[19];
  const float* dc_w2 = (const float*)d_in[20];
  const float* dc_b2 = (const float*)d_in[21];
  const float* g_w0  = (const float*)d_in[26];
  const float* g_b0  = (const float*)d_in[27];
  const float* g_w1  = (const float*)d_in[28];
  const float* g_b1  = (const float*)d_in[29];
  float* out = (float*)d_out;

  const int wgs = (TILES + WAVES_PER_WG - 1) / WAVES_PER_WG;  // 157
  ncde_rk4_wmma_kernel<<<wgs, THREADS, SMEM_TOTAL, stream>>>(
      ts, xc, x0,
      ez_w0, ez_b0, ez_w1, ez_b1, ez_w2, ez_b2,
      dc_w0, dc_b0, dc_w1, dc_b1, dc_w2, dc_b2,
      g_w0, g_b0, g_w1, g_b1, out);
}
